// ConditionalRandomField_89008902242642
// MI455X (gfx1250) — matrix operand order repaired and checked by reading, not compiled, not run on hardware
//
#include <hip/hip_runtime.h>
#include <hip/hip_bf16.h>

// CRF log-likelihood for B=64, T=1024, K=256 on gfx1250 (MI455X).
//
// Forward algorithm = sequential scan of log-semiring matmuls. Each step is a
// real f16 WMMA matmul against E = exp(transitions) (constant, register-
// resident B-fragments), stabilized by a per-row running max.
//
// State: alpha[b,:] = rel[b,:] + off[b], with off[b] a per-wave REGISTER.
// Per step (wave w owns row w in phase 1, a 16-col N-tile in phases 2/3):
//   m    = max_j rel[w,j]                 (wave-local shuffle reduce)
//   pA   = exp(rel[w,:] - m) (f16, LDS)   ; off[w] += m
//   acc  = pA(16x256) @ E(256x16)         (8x v_wmma_f32_16x16x32_f16,
//                                          A-fragments pinned in VGPRs so all
//                                          16 ds_load_b128 issue before one
//                                          s_wait_dscnt and the WMMAs chain)
//   rel' = log(acc) + logits[t] (+ end)   (branchless while t < min length)
// max_j pA = 1 and E >= ~0.7 => acc >= ~0.7: raw v_log_f32/v_exp_f32 safe.
//
// Mask note: CRF masks are prefix masks (reference derives end_idx =
// sum(mask)-1), so mask[b][t] != 0 <=> t < lens[b]. The scan splits at
// Lmin = min_b lens[b]: t < Lmin is fully branchless; the predicated tail
// is empty for the all-ones mask.

typedef _Float16 v8h  __attribute__((ext_vector_type(8)));
typedef _Float16 v16h __attribute__((ext_vector_type(16)));
typedef float    v8f  __attribute__((ext_vector_type(8)));

#define CRF_B 64
#define CRF_T 1024
#define CRF_K 256

__device__ __forceinline__ float fast_exp(float x) {
  return __builtin_amdgcn_exp2f(x * 1.44269504f);   // v_exp_f32
}
__device__ __forceinline__ float fast_log(float x) {
  return 0.69314718f * __builtin_amdgcn_logf(x);    // v_log_f32
}

// ---------------------------------------------------------------------------
// Prep: Ecol[j*K + i] = exp(transitions[i][j]) as f16, column-major so a WMMA
// B-fragment (col j, K-run of 16) is one contiguous 32B load. Also zero d_out.
// ---------------------------------------------------------------------------
__global__ void crf_prep_kernel(const float* __restrict__ trans,
                                _Float16* __restrict__ Ecol,
                                float* __restrict__ out) {
  const int i = blockIdx.x;   // source tag (K dim of the matmul)
  const int j = threadIdx.x;  // dest tag  (N dim)
  Ecol[(size_t)j * CRF_K + i] = (_Float16)fast_exp(trans[(size_t)i * CRF_K + j]);
  if (i == 0 && j == 0) out[0] = 0.0f;
}

// ---------------------------------------------------------------------------
// Numerator: gold-path gather/sum. One block per batch, 256 threads strided
// over T, wave32 shuffle + LDS block reduction (fully parallel gathers).
// ---------------------------------------------------------------------------
__global__ __launch_bounds__(256) void crf_numerator_kernel(
    const float* __restrict__ inputs, const int* __restrict__ tags,
    const int* __restrict__ mask, const float* __restrict__ trans,
    const float* __restrict__ startT, const float* __restrict__ endT,
    float* __restrict__ num) {
  const int b    = blockIdx.x;
  const int tid  = threadIdx.x;
  const int wave = tid >> 5, lane = tid & 31;
  const int* tg = tags + (size_t)b * CRF_T;
  const int* mk = mask + (size_t)b * CRF_T;
  const float* in = inputs + (size_t)b * CRF_T * CRF_K;

  float acc = 0.0f;
  int   lcl = 0;
  for (int t = tid; t < CRF_T; t += 256) {
    const int cur = tg[t];
    lcl += mk[t];
    if (t == 0) {
      acc += in[cur] + startT[cur];
    } else {
      acc += (float)mk[t] * (in[(size_t)t * CRF_K + cur] +
                             trans[(size_t)tg[t - 1] * CRF_K + cur]);
    }
  }
  for (int off = 16; off; off >>= 1) {
    acc += __shfl_xor(acc, off, 32);
    lcl += __shfl_xor(lcl, off, 32);
  }
  __shared__ float sacc[8];
  __shared__ int   slen[8];
  if (lane == 0) { sacc[wave] = acc; slen[wave] = lcl; }
  __syncthreads();
  if (tid == 0) {
    float a = 0.0f; int len = 0;
    for (int w = 0; w < 8; ++w) { a += sacc[w]; len += slen[w]; }
    num[b] = a + endT[tg[len - 1]];
  }
}

// ---------------------------------------------------------------------------
// Forward scan + denominator. Grid: 4 blocks x 512 threads (16 waves).
// Block bx owns batches [16*bx, 16*bx+16). Wave w owns output columns
// [16*w, 16*w+16). E B-fragments are register-resident for the whole scan.
// ---------------------------------------------------------------------------
__global__ __launch_bounds__(512) void crf_forward_kernel(
    const float* __restrict__ inputs, const int* __restrict__ mask,
    const float* __restrict__ startT, const float* __restrict__ endT,
    const _Float16* __restrict__ Ecol, const float* __restrict__ num,
    float* __restrict__ out) {
  const int b0   = blockIdx.x * 16;
  const int tid  = threadIdx.x;
  const int wave = tid >> 5;   // 0..15 : row owner (phase 1) / N-tile (2,3)
  const int lane = tid & 31;
  const int g    = lane >> 4;  // half-wave group (ISA fragment layouts)
  const int n    = lane & 15;
  const int j0   = wave * 16;

  __shared__ float    rel[16 * CRF_K];   // alpha - off[row], loop-carried
  __shared__ _Float16 pA[16 * CRF_K];    // exp(rel - rowmax)
  __shared__ int      lens[16];

  // Sequence lengths (wave w -> batch b0+w), wave32 shuffle reduction.
  {
    int s = 0;
    for (int t = lane; t < CRF_T; t += 32) s += mask[(size_t)(b0 + wave) * CRF_T + t];
    for (int off = 16; off; off >>= 1) s += __shfl_xor(s, off, 32);
    if (lane == 0) lens[wave] = s;
  }

  // B-fragments: E columns j0..j0+15, all 8 K-chunks, kept in VGPRs.
  // 32x16 f16 B in contiguous k order (same as A): lanes 0-15 hold N=lane, K=0..15; lanes 16-31 K=16..31.
  v16h bfrag[8];
#pragma unroll
  for (int kk = 0; kk < 8; ++kk)
    bfrag[kk] = *(const v16h*)(Ecol + (size_t)(j0 + n) * CRF_K + kk * 32 + g * 16);

  const float endj = endT[j0 + n];   // loop-invariant end-transition

  __syncthreads();

  // Hoist lengths into registers (own row + the 8 rows this lane updates).
  const int Lw = lens[wave];
  int Lr[8];
#pragma unroll
  for (int r = 0; r < 8; ++r) Lr[r] = lens[r + g * 8];
  int Lmin = lens[0];
  for (int i = 1; i < 16; ++i) Lmin = min(Lmin, lens[i]);

  // rel0 = logits[0] + start (+ end if length==1); off = 0. Wave w fills row w.
  float off_w = 0.0f;
  {
    const float ie0 = (Lw == 1) ? 1.0f : 0.0f;
#pragma unroll
    for (int e = 0; e < 8; ++e) {
      const int j = lane * 8 + e;
      rel[wave * CRF_K + j] =
          inputs[(size_t)(b0 + wave) * CRF_T * CRF_K + j] + startT[j] + ie0 * endT[j];
    }
  }
  __syncthreads();

  // Per-lane A-fragment LDS base (row M = lane&15) and logits base.
  const _Float16* aRow = pA + n * CRF_K;
#define LOAD_AFRAG(dst, kk)                                                   \
  {                                                                           \
    const v8h a0_ = *(const v8h*)(aRow + (kk) * 32 + g * 16);                 \
    const v8h a1_ = *(const v8h*)(aRow + (kk) * 32 + g * 16 + 8);             \
    dst = __builtin_shufflevector(a0_, a1_, 0, 1, 2, 3, 4, 5, 6, 7,           \
                                  8, 9, 10, 11, 12, 13, 14, 15);              \
  }

  // Software-pipelined logits: lg holds step t's values, loaded one step early.
  float lg[8];
#pragma unroll
  for (int r = 0; r < 8; ++r)
    lg[r] = inputs[((size_t)(b0 + r + g * 8) * CRF_T + 1) * CRF_K + j0 + n];

  // ---------------- Fast branchless scan: every row active -----------------
  for (int t = 1; t < Lmin; ++t) {
    // Phase 1: row max + pA = exp(rel - m), single contiguous row read.
    float vals[8];
#pragma unroll
    for (int e = 0; e < 8; ++e) vals[e] = rel[wave * CRF_K + lane * 8 + e];
    float m = vals[0];
#pragma unroll
    for (int e = 1; e < 8; ++e) m = fmaxf(m, vals[e]);
    for (int off = 16; off; off >>= 1) m = fmaxf(m, __shfl_xor(m, off, 32));
#pragma unroll
    for (int e = 0; e < 8; ++e)
      pA[wave * CRF_K + lane * 8 + e] = (_Float16)fast_exp(vals[e] - m);
    off_w += m;
    __syncthreads();

    // Phase 2: load all 8 A-fragments into pinned registers (16 pipelined
    // ds_load_b128, ONE s_wait_dscnt), then chain the 8 WMMAs back-to-back.
    v16h a0, a1, a2, a3, a4, a5, a6, a7;
    LOAD_AFRAG(a0, 0) LOAD_AFRAG(a1, 1) LOAD_AFRAG(a2, 2) LOAD_AFRAG(a3, 3)
    LOAD_AFRAG(a4, 4) LOAD_AFRAG(a5, 5) LOAD_AFRAG(a6, 6) LOAD_AFRAG(a7, 7)
    asm volatile("" : "+v"(a0), "+v"(a1), "+v"(a2), "+v"(a3),
                      "+v"(a4), "+v"(a5), "+v"(a6), "+v"(a7));
    v8f acc = {0.f, 0.f, 0.f, 0.f, 0.f, 0.f, 0.f, 0.f};
    acc = __builtin_amdgcn_wmma_f32_16x16x32_f16(false, a0, false, bfrag[0], (short)0, acc, false, false);
    acc = __builtin_amdgcn_wmma_f32_16x16x32_f16(false, a1, false, bfrag[1], (short)0, acc, false, false);
    acc = __builtin_amdgcn_wmma_f32_16x16x32_f16(false, a2, false, bfrag[2], (short)0, acc, false, false);
    acc = __builtin_amdgcn_wmma_f32_16x16x32_f16(false, a3, false, bfrag[3], (short)0, acc, false, false);
    acc = __builtin_amdgcn_wmma_f32_16x16x32_f16(false, a4, false, bfrag[4], (short)0, acc, false, false);
    acc = __builtin_amdgcn_wmma_f32_16x16x32_f16(false, a5, false, bfrag[5], (short)0, acc, false, false);
    acc = __builtin_amdgcn_wmma_f32_16x16x32_f16(false, a6, false, bfrag[6], (short)0, acc, false, false);
    acc = __builtin_amdgcn_wmma_f32_16x16x32_f16(false, a7, false, bfrag[7], (short)0, acc, false, false);

    // Pipeline: issue next step's logits loads now; they land under the
    // next barrier + phase 1 + WMMA chain. Distance-2 prefetch warms WGP$.
    float lgn[8];
#pragma unroll
    for (int r = 0; r < 8; ++r) {
      const size_t base = (size_t)(b0 + r + g * 8) * CRF_T;
      if (t + 1 < CRF_T) lgn[r] = inputs[(base + t + 1) * CRF_K + j0 + n];
      if (t + 2 < CRF_T)
        __builtin_prefetch(&inputs[(base + t + 2) * CRF_K + j0 + n], 0, 3);
    }

    // Phase 3: branchless rel update (all rows active; store unconditional).
    // ISA C layout: VGPR r -> row M = r + 8*g, col N = lane&15.
#pragma unroll
    for (int r = 0; r < 8; ++r) {
      const float ie = (t == Lr[r] - 1) ? endj : 0.0f;
      rel[(r + g * 8) * CRF_K + j0 + n] = fast_log(acc[r]) + lg[r] + ie;
    }
    __syncthreads();  // orders rel writes (and pA reads) vs next iteration
#pragma unroll
    for (int r = 0; r < 8; ++r) lg[r] = lgn[r];
  }

  // ---------------- Predicated tail (empty for all-ones mask) --------------
  for (int t = (Lmin > 1 ? Lmin : 1); t < CRF_T; ++t) {
    float vals[8];
#pragma unroll
    for (int e = 0; e < 8; ++e) vals[e] = rel[wave * CRF_K + lane * 8 + e];
    float m = vals[0];
#pragma unroll
    for (int e = 1; e < 8; ++e) m = fmaxf(m, vals[e]);
    for (int off = 16; off; off >>= 1) m = fmaxf(m, __shfl_xor(m, off, 32));
#pragma unroll
    for (int e = 0; e < 8; ++e)
      pA[wave * CRF_K + lane * 8 + e] = (_Float16)fast_exp(vals[e] - m);
    if (t < Lw) off_w += m;
    __syncthreads();

    float lgt[8];
#pragma unroll
    for (int r = 0; r < 8; ++r)
      lgt[r] = inputs[((size_t)(b0 + r + g * 8) * CRF_T + t) * CRF_K + j0 + n];

    v16h a0, a1, a2, a3, a4, a5, a6, a7;
    LOAD_AFRAG(a0, 0) LOAD_AFRAG(a1, 1) LOAD_AFRAG(a2, 2) LOAD_AFRAG(a3, 3)
    LOAD_AFRAG(a4, 4) LOAD_AFRAG(a5, 5) LOAD_AFRAG(a6, 6) LOAD_AFRAG(a7, 7)
    asm volatile("" : "+v"(a0), "+v"(a1), "+v"(a2), "+v"(a3),
                      "+v"(a4), "+v"(a5), "+v"(a6), "+v"(a7));
    v8f acc = {0.f, 0.f, 0.f, 0.f, 0.f, 0.f, 0.f, 0.f};
    acc = __builtin_amdgcn_wmma_f32_16x16x32_f16(false, a0, false, bfrag[0], (short)0, acc, false, false);
    acc = __builtin_amdgcn_wmma_f32_16x16x32_f16(false, a1, false, bfrag[1], (short)0, acc, false, false);
    acc = __builtin_amdgcn_wmma_f32_16x16x32_f16(false, a2, false, bfrag[2], (short)0, acc, false, false);
    acc = __builtin_amdgcn_wmma_f32_16x16x32_f16(false, a3, false, bfrag[3], (short)0, acc, false, false);
    acc = __builtin_amdgcn_wmma_f32_16x16x32_f16(false, a4, false, bfrag[4], (short)0, acc, false, false);
    acc = __builtin_amdgcn_wmma_f32_16x16x32_f16(false, a5, false, bfrag[5], (short)0, acc, false, false);
    acc = __builtin_amdgcn_wmma_f32_16x16x32_f16(false, a6, false, bfrag[6], (short)0, acc, false, false);
    acc = __builtin_amdgcn_wmma_f32_16x16x32_f16(false, a7, false, bfrag[7], (short)0, acc, false, false);

#pragma unroll
    for (int r = 0; r < 8; ++r) {
      const float ie = (t == Lr[r] - 1) ? endj : 0.0f;
      const float nv = fast_log(acc[r]) + lgt[r] + ie;
      if (t < Lr[r]) rel[(r + g * 8) * CRF_K + j0 + n] = nv;
    }
    __syncthreads();
  }
#undef LOAD_AFRAG

  // denominator[b] = off[b] + logsumexp_j(rel[b,:]); accumulate num - denom.
  {
    float vals[8];
#pragma unroll
    for (int e = 0; e < 8; ++e) vals[e] = rel[wave * CRF_K + lane * 8 + e];
    float m = vals[0];
#pragma unroll
    for (int e = 1; e < 8; ++e) m = fmaxf(m, vals[e]);
    for (int off = 16; off; off >>= 1) m = fmaxf(m, __shfl_xor(m, off, 32));
    float s = 0.0f;
#pragma unroll
    for (int e = 0; e < 8; ++e) s += fast_exp(vals[e] - m);
    for (int off = 16; off; off >>= 1) s += __shfl_xor(s, off, 32);
    if (lane == 0) {
      const float denom = off_w + m + fast_log(s);
      atomicAdd(out, num[b0 + wave] - denom);
    }
  }
}

extern "C" void kernel_launch(void* const* d_in, const int* in_sizes, int n_in,
                              void* d_out, int out_size, void* d_ws, size_t ws_size,
                              hipStream_t stream) {
  (void)in_sizes; (void)n_in; (void)out_size; (void)ws_size;
  const float* inputs = (const float*)d_in[0];
  const int*   tags   = (const int*)d_in[1];   // jnp.int64 w/o x64 -> int32
  const int*   mask   = (const int*)d_in[2];
  const float* trans  = (const float*)d_in[3];
  const float* startT = (const float*)d_in[4];
  const float* endT   = (const float*)d_in[5];
  float* out = (float*)d_out;

  _Float16* Ecol = (_Float16*)d_ws;                       // 256*256*2 = 128 KB
  float*    num  = (float*)((char*)d_ws + (size_t)CRF_K * CRF_K * sizeof(_Float16));

  crf_prep_kernel<<<CRF_K, CRF_K, 0, stream>>>(trans, Ecol, out);
  crf_numerator_kernel<<<CRF_B, 256, 0, stream>>>(inputs, tags, mask, trans,
                                                  startT, endT, num);
  crf_forward_kernel<<<CRF_B / 16, 512, 0, stream>>>(inputs, mask, startT, endT,
                                                     Ecol, num, out);
}
